// Decoder_20246475833613
// MI455X (gfx1250) — matrix operand order repaired and checked by reading, not compiled, not run on hardware
//
#include <hip/hip_runtime.h>
#include <hip/hip_bf16.h>

// MI455X / gfx1250, wave32. Fused decoder:
//   prep:    W1 (131x128)      -> W1t f16 [128][160] col-major (K padded to 160)
//            W2 (128x32)       -> W2t f16 [32][128]  col-major
//            W_stretch (16x128)-> Wst f16 [128][32]  col-major (K padded to 32, zeros)
//   stretch: skin_fea f16 [8192][128] = qcf @ W_stretch + b  (WMMA)
//   main:    per wave: 16 rows of N -> weights -> gather/interp (pk f16 FMA)
//            -> WMMA GEMM1 (relu) -> WMMA GEMM2 (relu) -> tiny GEMM3 + residual
// ws usage: (8192*128 + 128*160 + 32*128 + 128*32) * 2 = 2,154,496 bytes.

typedef _Float16 v16h __attribute__((ext_vector_type(16)));
typedef _Float16 v8h  __attribute__((ext_vector_type(8)));
typedef float    v8f  __attribute__((ext_vector_type(8)));

#define EPSF 1e-8f

union FragA {
    v16h v;
    unsigned int u[8];
    v8h h8[2];
};

static __device__ __forceinline__ v8f wmma_f16(v16h a, v16h b, v8f c) {
    return __builtin_amdgcn_wmma_f32_16x16x32_f16(false, a, false, b, (short)0, c,
                                                  false, false);
}

// ---------------------------------------------------------------------------
// Weight transposes / f16 conversion (tiny)
// ---------------------------------------------------------------------------
__global__ void k_prep(const float* __restrict__ W1, const float* __restrict__ W2,
                       const float* __restrict__ Ws,
                       _Float16* __restrict__ W1t, _Float16* __restrict__ W2t,
                       _Float16* __restrict__ Wst) {
    int tid = blockIdx.x * blockDim.x + threadIdx.x;
    int nth = gridDim.x * blockDim.x;
    for (int i = tid; i < 128 * 160; i += nth) {
        int col = i / 160, k = i % 160;
        W1t[i] = (k < 131) ? (_Float16)W1[k * 128 + col] : (_Float16)0.f;
    }
    for (int i = tid; i < 32 * 128; i += nth) {
        int col = i / 128, k = i % 128;
        W2t[i] = (_Float16)W2[k * 32 + col];
    }
    for (int i = tid; i < 128 * 32; i += nth) {
        int col = i / 32, k = i % 32;
        Wst[i] = (k < 16) ? (_Float16)Ws[k * 128 + col] : (_Float16)0.f;
    }
}

// ---------------------------------------------------------------------------
// skin_fea = qcf(8192x16) @ W_stretch(16x128) + b_stretch  -> f16, via WMMA
// one wave per 16-row tile; 64 blocks x 8 waves = 512 tiles = 8192 rows
// ---------------------------------------------------------------------------
__global__ void __launch_bounds__(256) k_stretch(
        const float* __restrict__ qcf, const _Float16* __restrict__ Wst,
        const float* __restrict__ bs, _Float16* __restrict__ sf) {
    const int lane = threadIdx.x & 31;
    const int wave = blockIdx.x * 8 + (threadIdx.x >> 5);
    const int m0   = wave * 16;
    const int row  = lane & 15;
    const bool lo  = lane < 16;
    const int kg   = lo ? 0 : 8;
    const int kh   = lo ? 0 : 16;

    // A: row (m0+row), K = kg..kg+7 valid (contiguous 32B), K 16..31 zero pad
    FragA a;
    {
        float4 q0 = *(const float4*)(qcf + (m0 + row) * 16 + kg);
        float4 q1 = *(const float4*)(qcf + (m0 + row) * 16 + kg + 4);
        a.v[0] = (_Float16)q0.x; a.v[1] = (_Float16)q0.y;
        a.v[2] = (_Float16)q0.z; a.v[3] = (_Float16)q0.w;
        a.v[4] = (_Float16)q1.x; a.v[5] = (_Float16)q1.y;
        a.v[6] = (_Float16)q1.z; a.v[7] = (_Float16)q1.w;
#pragma unroll
        for (int i = 8; i < 16; ++i) a.v[i] = (_Float16)0.f;
    }

    for (int c = 0; c < 8; ++c) {
        int col = c * 16 + row;
        FragA b;   // unconditional contiguous loads; h8[1] (k 16..31) reads zero pad
        b.h8[0] = *(const v8h*)(Wst + col * 32 + kg);
        b.h8[1] = *(const v8h*)(Wst + col * 32 + kg + 16);
        v8f acc = {};
        acc = wmma_f16(a.v, b.v, acc);
        float bias = bs[col];
#pragma unroll
        for (int r = 0; r < 8; ++r) {
            int mm = m0 + r + (lo ? 0 : 8);
            sf[mm * 128 + col] = (_Float16)(acc[r] + bias);
        }
    }
}

// ---------------------------------------------------------------------------
// Fused main kernel: one wave handles 16 consecutive n rows.
// 1024 blocks x 256 threads (8 waves) -> 8192 tiles = 131072 rows.
// Dynamic LDS: W1t(40960) + W2t(8192) + 8 * 12288 per-wave = 147456 bytes.
// ---------------------------------------------------------------------------
__global__ void __launch_bounds__(256) k_main(
        const _Float16* __restrict__ sf, const int* __restrict__ d_idx,
        const float* __restrict__ dwin, const float* __restrict__ bones,
        const _Float16* __restrict__ W1t, const _Float16* __restrict__ W2t,
        const float* __restrict__ b1, const float* __restrict__ b2,
        const float* __restrict__ W3, const float* __restrict__ b3,
        float* __restrict__ out) {
    extern __shared__ char smem[];
    _Float16* W1s = (_Float16*)smem;              // [128][160]
    _Float16* W2s = W1s + 128 * 160;              // [32][128]
    char* pw = (char*)(W2s + 32 * 128) + (threadIdx.x >> 5) * 12288;
    _Float16* xbuf = (_Float16*)pw;               // [16][160] f16
    _Float16* hbuf = xbuf + 16 * 160;             // [16][128] f16
    float*    h2   = (float*)(hbuf + 16 * 128);   // [16][32]  f32
    float*    wgt  = h2 + 16 * 32;                // [16][8]
    int*      widx = (int*)(wgt + 16 * 8);        // [16][8]

    // ---- stage weights into LDS (whole block cooperates) ----
    {
        const uint4* s1 = (const uint4*)W1t;
        uint4* d1 = (uint4*)W1s;
        for (int i = threadIdx.x; i < (128 * 160) / 8; i += 256) d1[i] = s1[i];
        const uint4* s2 = (const uint4*)W2t;
        uint4* d2 = (uint4*)W2s;
        for (int i = threadIdx.x; i < (32 * 128) / 8; i += 256) d2[i] = s2[i];
    }
    __syncthreads();

    const int lane = threadIdx.x & 31;
    const int tile = blockIdx.x * 8 + (threadIdx.x >> 5);
    const int n0   = tile * 16;
    const int row  = lane & 15;
    const bool lo  = lane < 16;

    // ---- phase 1: inverse-distance weights ----
    float dwx = dwin[(n0 + row) * 3 + 0];
    float dwy = dwin[(n0 + row) * 3 + 1];
    float dwz = dwin[(n0 + row) * 3 + 2];
    {
        const int kb = lo ? 0 : 4;   // lanes 0-15: k 0..3, lanes 16-31: k 4..7
        float wv[4];
        int   id[4];
        float s = 0.f;
#pragma unroll
        for (int kk = 0; kk < 4; ++kk) {
            int k = kb + kk;
            int i = d_idx[(n0 + row) * 8 + k];
            float rx = dwx - bones[i * 3 + 0];
            float ry = dwy - bones[i * 3 + 1];
            float rz = dwz - bones[i * 3 + 2];
            float dist = sqrtf(rx * rx + ry * ry + rz * rz + EPSF);
            wv[kk] = 1.f / (dist + EPSF);
            id[kk] = i;
            s += wv[kk];
        }
        float tot = s + __shfl_xor(s, 16);   // reduce across lane-halves (wave32)
        float inv = 1.f / tot;
#pragma unroll
        for (int kk = 0; kk < 4; ++kk) {
            wgt[row * 8 + kb + kk]  = wv[kk] * inv;
            widx[row * 8 + kb + kk] = id[kk];
        }
    }
    asm volatile("s_wait_dscnt 0" ::: "memory");

    // ---- phase 2: gather + interp -> xbuf (f16, K padded to 160) ----
    // lanes 0-15: rows 0-7, lanes 16-31: rows 8-15; 8 f16 channels per lane.
    // Convex combination of 8 normalized weights -> packed f16 FMA is safe.
    {
        const int cb = (lane & 15) * 8;
        const int rbase = lo ? 0 : 8;
#pragma unroll
        for (int rr = 0; rr < 8; ++rr) {
            int r = rbase + rr;
            float4 wq0 = *(const float4*)(wgt + r * 8);
            float4 wq1 = *(const float4*)(wgt + r * 8 + 4);
            int4   iq0 = *(const int4*)(widx + r * 8);
            int4   iq1 = *(const int4*)(widx + r * 8 + 4);
            float wk[8] = {wq0.x, wq0.y, wq0.z, wq0.w, wq1.x, wq1.y, wq1.z, wq1.w};
            int   ik[8] = {iq0.x, iq0.y, iq0.z, iq0.w, iq1.x, iq1.y, iq1.z, iq1.w};
            v8h acc = {};
#pragma unroll
            for (int k = 0; k < 8; ++k) {
                v8h f = *(const v8h*)(sf + ik[k] * 128 + cb);  // b128, L2-resident
                _Float16 wh = (_Float16)wk[k];
                v8h ws = {wh, wh, wh, wh, wh, wh, wh, wh};
                acc += f * ws;                                 // 4x v_pk_fma_f16
            }
            *(v8h*)(xbuf + r * 160 + cb) = acc;                // one b128 ds store
        }
        // channels 128..130 = d_windows, 131..159 = 0 pad
        int chb = 128 + (lo ? 0 : 16);
#pragma unroll
        for (int j = 0; j < 16; ++j) {
            int ch = chb + j;
            float v = (ch == 128) ? dwx : (ch == 129) ? dwy : (ch == 130) ? dwz : 0.f;
            xbuf[row * 160 + ch] = (_Float16)v;
        }
    }
    asm volatile("s_wait_dscnt 0" ::: "memory");

    // ---- GEMM1: h = relu(x @ W1 + b1), K = 160, 8 col tiles ----
    FragA af[5];
    {
        const int kg = lo ? 0 : 8;
#pragma unroll
        for (int s5 = 0; s5 < 5; ++s5) {
#pragma unroll
            for (int p = 0; p < 8; ++p) {
                int K = s5 * 32 + ((p < 4) ? (kg + 2 * p) : (16 + kg + 2 * (p - 4)));
                af[s5].u[p] = *(const unsigned int*)(xbuf + row * 160 + K);
            }
        }
    }
    {
        const int kh = lo ? 0 : 16;
        for (int c = 0; c < 8; ++c) {
            int col = c * 16 + row;
            v8f acc = {};
#pragma unroll
            for (int s5 = 0; s5 < 5; ++s5) {
                FragA b;
                b.h8[0] = *(const v8h*)(W1s + col * 160 + s5 * 32 + kh / 2);
                b.h8[1] = *(const v8h*)(W1s + col * 160 + s5 * 32 + kh / 2 + 16);
                acc = wmma_f16(af[s5].v, b.v, acc);
            }
            float bias = b1[col];
#pragma unroll
            for (int r = 0; r < 8; ++r) {
                float v = acc[r] + bias;
                v = v > 0.f ? v : 0.f;
                int mm = r + (lo ? 0 : 8);
                hbuf[mm * 128 + col] = (_Float16)v;
            }
        }
    }
    asm volatile("s_wait_dscnt 0" ::: "memory");

    // ---- GEMM2: h2 = relu(h @ W2 + b2), K = 128, 2 col tiles ----
    FragA ah[4];
    {
        const int kg = lo ? 0 : 8;
#pragma unroll
        for (int s4 = 0; s4 < 4; ++s4) {
#pragma unroll
            for (int p = 0; p < 8; ++p) {
                int K = s4 * 32 + ((p < 4) ? (kg + 2 * p) : (16 + kg + 2 * (p - 4)));
                ah[s4].u[p] = *(const unsigned int*)(hbuf + row * 128 + K);
            }
        }
    }
    {
        const int kh = lo ? 0 : 16;
#pragma unroll
        for (int c = 0; c < 2; ++c) {
            int col = c * 16 + row;
            v8f acc = {};
#pragma unroll
            for (int s4 = 0; s4 < 4; ++s4) {
                FragA b;
                b.h8[0] = *(const v8h*)(W2s + col * 128 + s4 * 32 + kh / 2);
                b.h8[1] = *(const v8h*)(W2s + col * 128 + s4 * 32 + kh / 2 + 16);
                acc = wmma_f16(ah[s4].v, b.v, acc);
            }
            float bias = b2[col];
#pragma unroll
            for (int r = 0; r < 8; ++r) {
                float v = acc[r] + bias;
                v = v > 0.f ? v : 0.f;
                int mm = r + (lo ? 0 : 8);
                h2[mm * 32 + col] = v;
            }
        }
    }
    asm volatile("s_wait_dscnt 0" ::: "memory");

    // ---- GEMM3 (8x3) + residual, 64 (row, r_up) tasks over 32 lanes ----
    float w3[8][3], bb[3];
#pragma unroll
    for (int f = 0; f < 8; ++f)
#pragma unroll
        for (int o = 0; o < 3; ++o) w3[f][o] = W3[f * 3 + o];
#pragma unroll
    for (int o = 0; o < 3; ++o) bb[o] = b3[o];

#pragma unroll
    for (int it = 0; it < 2; ++it) {
        int task = lane + 32 * it;
        int rr = task >> 2;   // row in tile
        int ru = task & 3;    // r_up
        float hh[8];
#pragma unroll
        for (int f = 0; f < 8; ++f) hh[f] = h2[rr * 32 + ru * 8 + f];
        float ox = bb[0], oy = bb[1], oz = bb[2];
#pragma unroll
        for (int f = 0; f < 8; ++f) {
            ox += hh[f] * w3[f][0];
            oy += hh[f] * w3[f][1];
            oz += hh[f] * w3[f][2];
        }
        int n = n0 + rr;
        float bx = dwin[n * 3 + 0], by = dwin[n * 3 + 1], bz = dwin[n * 3 + 2];
        int base = (n * 4 + ru) * 3;
        out[base + 0] = bx + ox;
        out[base + 1] = by + oy;
        out[base + 2] = bz + oz;
    }
}

// ---------------------------------------------------------------------------
extern "C" void kernel_launch(void* const* d_in, const int* in_sizes, int n_in,
                              void* d_out, int out_size, void* d_ws, size_t ws_size,
                              hipStream_t stream) {
    (void)in_sizes; (void)n_in; (void)out_size; (void)ws_size;
    const float* qcf   = (const float*)d_in[0];
    const int*   didx  = (const int*)d_in[1];
    const float* dwin  = (const float*)d_in[2];
    const float* bones = (const float*)d_in[3];
    // d_in[4] = K scalar (always 8)
    const float* Ws = (const float*)d_in[5];
    const float* bs = (const float*)d_in[6];
    const float* W1 = (const float*)d_in[7];
    const float* b1 = (const float*)d_in[8];
    const float* W2 = (const float*)d_in[9];
    const float* b2 = (const float*)d_in[10];
    const float* W3 = (const float*)d_in[11];
    const float* b3 = (const float*)d_in[12];
    float* out = (float*)d_out;

    _Float16* sf  = (_Float16*)d_ws;          // 8192*128 f16
    _Float16* W1t = sf + 8192 * 128;          // 128*160 f16
    _Float16* W2t = W1t + 128 * 160;          // 32*128 f16
    _Float16* Wst = W2t + 32 * 128;           // 128*32 f16 (K-padded, zeros)

    k_prep<<<24, 256, 0, stream>>>(W1, W2, Ws, W1t, W2t, Wst);
    k_stretch<<<64, 256, 0, stream>>>(qcf, Wst, bs, sf);
    const size_t lds = 128 * 160 * 2 + 32 * 128 * 2 + 8 * 12288; // 147456 B
    k_main<<<1024, 256, lds, stream>>>(sf, didx, dwin, bones, W1t, W2t,
                                       b1, b2, W3, b3, out);
}
